// QuantizedLinear_34754875359403
// MI455X (gfx1250) — matrix operand order repaired and checked by reading, not compiled, not run on hardware
//
#include <hip/hip_runtime.h>

// ---- CDNA5 (gfx1250, wave32) Q4_0 quantized GEMM via v_wmma_f32_16x16x32_f16 ----
// out[M=8192, N=11008] = x[M, K=4096] * W^T + bias, W stored Q4_0 (18B / 32 elems).
// Software-pipelined: global->reg prefetch of tile t+1 overlaps the 16 WMMAs of
// tile t; LDS double-buffered (72 KB of the 320 KB WGP budget), 1 barrier/tile.
// Dequant uses the exact f16 bit trick: (0x6400|nib) = 1024+nib, minus 1032 gives
// nib-8 exactly, then one f16 multiply by the scale (bit-identical to reference).

typedef _Float16 v16h __attribute__((ext_vector_type(16)));
typedef _Float16 h8   __attribute__((ext_vector_type(8)));
typedef _Float16 h4   __attribute__((ext_vector_type(4)));
typedef _Float16 h2   __attribute__((ext_vector_type(2)));
typedef float    v8f  __attribute__((ext_vector_type(8)));
typedef float    v4f  __attribute__((ext_vector_type(4)));

namespace {
constexpr int K_IN   = 4096;   // IN_F
constexpr int N_OUT  = 11008;  // OUT_F
constexpr int M_TOT  = 4 * 2048;

constexpr int BM = 128;        // workgroup M tile
constexpr int BN = 128;        // workgroup N tile
constexpr int BK = 64;         // K staged per LDS round (2 wmma k-steps)
constexpr int LDT = 72;        // padded LDS row stride in f16 (144B = 36 dwords)
constexpr int NTHREADS = 256;  // 8 wave32 waves
constexpr int NTILES = K_IN / BK; // 64
}

__global__ __launch_bounds__(NTHREADS)
void q4_0_gemm_wmma(const float* __restrict__ x,
                    const unsigned short* __restrict__ w16, // w_quant viewed as u16
                    const float* __restrict__ bias,
                    float* __restrict__ out)
{
    __shared__ _Float16 sA[2][BM * LDT];  // [buf][m][k], 2 x 18 KB
    __shared__ _Float16 sB[2][BN * LDT];  // [buf][n][k], 2 x 18 KB

    const int tid   = threadIdx.x;
    const int lane  = tid & 31;
    const int wid   = tid >> 5;        // 0..7
    const int lrow  = lane & 15;       // row/col within 16x16 frag
    const int lhalf = lane >> 4;       // which half-wave

    const int mBase = blockIdx.y * BM;
    const int nBase = blockIdx.x * BN;
    const int wm = (wid & 1) * 64;     // wave M origin in tile (2 waves along M)
    const int wn = (wid >> 1) * 32;    // wave N origin in tile (4 waves along N)

    // Per-thread staging assignments (constant across tiles)
    const int bN  = tid >> 1;          // B: output channel within tile, 0..127
    const int bKB = tid & 1;           // B: which 32-k block of the BK chunk

    v8f acc[4][2];
#pragma unroll
    for (int i = 0; i < 4; ++i)
#pragma unroll
        for (int j = 0; j < 2; ++j)
            acc[i][j] = {};

    // Prefetch registers for the next K-tile
    v4f            areg[8];            // A: 8 float4 per thread (128x64 fp32 tile)
    unsigned short breg[9];            // B: one Q4_0 block (scale + 16 nibble bytes)

    auto fetch = [&](int kt) {
#pragma unroll
        for (int it = 0; it < 8; ++it) {
            const int f   = tid + it * NTHREADS;   // float4 index, 0..2047
            const int row = f >> 4;                // 0..127
            const int c4  = (f & 15) * 4;          // k offset 0..60
            areg[it] = *(const v4f*)(x + (size_t)(mBase + row) * K_IN + kt + c4);
        }
        const int gb = (nBase + bN) * (K_IN / 32) + (kt >> 5) + bKB;
        const unsigned short* bp = w16 + (size_t)gb * 9;  // 18B block, u16-aligned
#pragma unroll
        for (int j = 0; j < 9; ++j) breg[j] = bp[j];
    };

    auto stage = [&](int buf) {
        // A: fp32 -> f16 into LDS
#pragma unroll
        for (int it = 0; it < 8; ++it) {
            const int f   = tid + it * NTHREADS;
            const int row = f >> 4;
            const int c4  = (f & 15) * 4;
            const h4 hv = { (_Float16)areg[it][0], (_Float16)areg[it][1],
                            (_Float16)areg[it][2], (_Float16)areg[it][3] };
            *(h4*)(&sA[buf][row * LDT + c4]) = hv;
        }
        // B: dequant one Q4_0 block into LDS (exact f16 bit trick, packed math)
        _Float16 sh;
        __builtin_memcpy(&sh, &breg[0], sizeof(sh));
        const h2 s2   = { sh, sh };
        const h2 off2 = { (_Float16)(-1032.0f), (_Float16)(-1032.0f) };
        _Float16* dst = &sB[buf][bN * LDT + bKB * 32];
#pragma unroll
        for (int j = 0; j < 8; ++j) {
            const unsigned int q = breg[1 + j];      // bytes b0 (lo), b1 (hi)
            // {lo(b0), hi(b0)} and {lo(b1), hi(b1)} as f16 = 1024 + nib
            const unsigned int p01 = (q & 0xFu)        | ((q & 0xF0u)   << 12) | 0x64006400u;
            const unsigned int p23 = ((q >> 8) & 0xFu) | ((q & 0xF000u) << 4)  | 0x64006400u;
            // (1024+nib) - 1032 = nib-8 (exact), then one rounding multiply by scale
            const h2 r01 = (__builtin_bit_cast(h2, p01) + off2) * s2;
            const h2 r23 = (__builtin_bit_cast(h2, p23) + off2) * s2;
            const h2 lo = { r01[0], r23[0] };
            const h2 hi = { r01[1], r23[1] };
            *(h2*)(dst + 2 * j) = lo;
            *(h2*)(dst + 16 + 2 * j) = hi;
        }
    };

    auto compute = [&](int buf) {
#pragma unroll
        for (int kk = 0; kk < BK; kk += 32) {
            v16h af[4], bf[2];
#pragma unroll
            for (int i = 0; i < 4; ++i) {
                // A 16x32 f16 layout: lane half selects k run {0..7} vs {8..15},
                // second 16 k in elements 8..15 -> two contiguous 16B LDS loads.
                const _Float16* p = &sA[buf][(wm + 16 * i + lrow) * LDT + kk + 8 * lhalf];
                ((h8*)&af[i])[0] = *(const h8*)p;
                ((h8*)&af[i])[1] = *(const h8*)(p + 16);
            }
#pragma unroll
            for (int j = 0; j < 2; ++j) {
                // B 32x16 f16 layout: lane = column, same k runs as A.
                const _Float16* p = &sB[buf][(wn + 16 * j + lrow) * LDT + kk + 8 * lhalf];
                ((h8*)&bf[j])[0] = *(const h8*)p;
                ((h8*)&bf[j])[1] = *(const h8*)(p + 16);
            }
#pragma unroll
            for (int i = 0; i < 4; ++i)
#pragma unroll
                for (int j = 0; j < 2; ++j)
                    acc[i][j] = __builtin_amdgcn_wmma_f32_16x16x32_f16(
                        /*neg_a=*/false, af[i], /*neg_b=*/false, bf[j],
                        /*c_mod=*/(short)0, acc[i][j],
                        /*reuse_a=*/false, /*reuse_b=*/false);
        }
    };

    // ---- Software pipeline: prologue stages tile 0 ----
    fetch(0);
    stage(0);
    __syncthreads();

    for (int t = 0; t < NTILES; ++t) {
        const int cur = t & 1;
        if (t + 1 < NTILES) fetch((t + 1) * BK);  // global loads in flight under WMMAs
        compute(cur);
        if (t + 1 < NTILES) stage(1 - cur);       // publish next buffer
        __syncthreads();                          // retire reads of cur + publish 1-cur
    }

    // ---- Epilogue: C/D layout -> lane L, reg r holds (M = 8*(L>=16)+r, N = L&15) ----
#pragma unroll
    for (int j = 0; j < 2; ++j) {
        const int n  = nBase + wn + 16 * j + lrow;
        const float bv = bias[n];
#pragma unroll
        for (int i = 0; i < 4; ++i) {
            const int mrow = mBase + wm + 16 * i + 8 * lhalf;
#pragma unroll
            for (int r = 0; r < 8; ++r) {
                out[(size_t)(mrow + r) * N_OUT + n] = acc[i][j][r] + bv;
            }
        }
    }
}

extern "C" void kernel_launch(void* const* d_in, const int* in_sizes, int n_in,
                              void* d_out, int out_size, void* d_ws, size_t ws_size,
                              hipStream_t stream) {
    (void)in_sizes; (void)n_in; (void)d_ws; (void)ws_size; (void)out_size;
    const float*          xp = (const float*)d_in[0];
    const unsigned short* wp = (const unsigned short*)d_in[1]; // u8 buffer, 2B aligned
    const float*          bp = (const float*)d_in[2];
    float*                op = (float*)d_out;

    dim3 grid(N_OUT / BN, M_TOT / BM);  // (86, 64)
    dim3 block(NTHREADS);
    q4_0_gemm_wmma<<<grid, block, 0, stream>>>(xp, wp, bp, op);
}
